// CoarseMatching_90838558310961
// MI455X (gfx1250) — matrix operand order repaired and checked by reading, not compiled, not run on hardware
//
#include <hip/hip_runtime.h>

typedef _Float16 v8h  __attribute__((ext_vector_type(8)));
typedef _Float16 v16h __attribute__((ext_vector_type(16)));
typedef float    v8f  __attribute__((ext_vector_type(8)));

#define THR_C    0.2f
#define BORDER_C 2

// -------------------------------------------------------------------------
// 1) fp32 -> f16 with pre-scale (1/16 per operand so dot = raw/256 = scaled)
// -------------------------------------------------------------------------
__global__ void cvt_f16_scale(const float* __restrict__ in,
                              _Float16* __restrict__ out, int n, float scale) {
  int i = blockIdx.x * blockDim.x + threadIdx.x;
  if (i < n) out[i] = (_Float16)(in[i] * scale);
}

// -------------------------------------------------------------------------
// 2) sim = A(f16) x B(f16)^T * postScale  via v_wmma_f32_16x16x32_f16
//    block tile 128x128, 8 waves, wave tile 64x32 (4x2 subtiles of 16x16).
//    Double-buffered LDS, branch-free staging (OOB rows/cols clamped: they
//    only feed output elements the epilogue never stores).
// -------------------------------------------------------------------------
__launch_bounds__(256)
__global__ void sim_gemm_wmma(const _Float16* __restrict__ A,  // [N][L][C]
                              const _Float16* __restrict__ B,  // [N][S][C]
                              float* __restrict__ simOut,      // [N][L][S]
                              int L, int S, float postScale) {
  const int BK = 32;
  const int C = 256;
  const int KITERS = C / BK;  // 8
  __shared__ __attribute__((aligned(64))) _Float16 As[2][128][32];  // 16 KB
  __shared__ __attribute__((aligned(64))) _Float16 Bs[2][128][32];  // 16 KB

  const int nb      = blockIdx.z;
  const int rowBase = blockIdx.y * 128;
  const int colBase = blockIdx.x * 128;
  const int tid  = threadIdx.x;
  const int lane = tid & 31;
  const int wave = tid >> 5;
  const int waveM = wave >> 2;   // 0..1 -> 64 rows each
  const int waveN = wave & 3;    // 0..3 -> 32 cols each
  const int lm = lane & 15;
  const int hs = lane >> 4;      // half-wave select

  const _Float16* Ag = A + (size_t)nb * L * C;
  const _Float16* Bg = B + (size_t)nb * S * C;

  // Per-thread staging: 512 16B chunks per matrix / 256 threads = 2 chunks each.
  const int r0 = tid >> 2;                 // 0..63
  const int r1 = r0 + 64;                  // 64..127
  const int o8 = (tid & 3) * 8;            // half offset 0,8,16,24
  const int ar0 = min(rowBase + r0, L - 1);
  const int ar1 = min(rowBase + r1, L - 1);
  const int bc0 = min(colBase + r0, S - 1);
  const int bc1 = min(colBase + r1, S - 1);
  const _Float16* ag0 = Ag + (size_t)ar0 * C + o8;
  const _Float16* ag1 = Ag + (size_t)ar1 * C + o8;
  const _Float16* bg0 = Bg + (size_t)bc0 * C + o8;
  const _Float16* bg1 = Bg + (size_t)bc1 * C + o8;

  // ---- prologue: stage K-slab 0 into buffer 0 ----
  {
    uint4 va0 = *(const uint4*)(ag0);
    uint4 va1 = *(const uint4*)(ag1);
    uint4 vb0 = *(const uint4*)(bg0);
    uint4 vb1 = *(const uint4*)(bg1);
    *(uint4*)&As[0][r0][o8] = va0;
    *(uint4*)&As[0][r1][o8] = va1;
    *(uint4*)&Bs[0][r0][o8] = vb0;
    *(uint4*)&Bs[0][r1][o8] = vb1;
  }

  v8f acc[4][2];
#pragma unroll
  for (int i = 0; i < 4; ++i)
#pragma unroll
    for (int j = 0; j < 2; ++j)
#pragma unroll
      for (int e = 0; e < 8; ++e) acc[i][j][e] = 0.0f;

#pragma unroll
  for (int ki = 0; ki < KITERS; ++ki) {
    const int cur = ki & 1;
    const int nxt = cur ^ 1;
    const int kn  = (ki + 1) * BK;
    __syncthreads();  // buffer `cur` ready; buffer `nxt` free

    // ---- fragment loads from LDS buffer `cur` (ISA wave32 layouts) ----
    v16h aF[4], bF[2];
#pragma unroll
    for (int tm = 0; tm < 4; ++tm) {
      int r = waveM * 64 + tm * 16 + lm;
      v8h lo = *(const v8h*)&As[cur][r][hs * 8];       // K 0..7  / 8..15
      v8h hi = *(const v8h*)&As[cur][r][hs * 8 + 16];  // K 16..23 / 24..31
#pragma unroll
      for (int i = 0; i < 8; ++i) { aF[tm][i] = lo[i]; aF[tm][i + 8] = hi[i]; }
    }
#pragma unroll
    for (int tn = 0; tn < 2; ++tn) {
      int cc = waveN * 32 + tn * 16 + lm;              // output column
      v8h blo = *(const v8h*)&Bs[cur][cc][hs * 8];       // K 0..7  / 8..15
      v8h bhi = *(const v8h*)&Bs[cur][cc][hs * 8 + 16];  // K 16..23 / 24..31
#pragma unroll
      for (int i = 0; i < 8; ++i) { bF[tn][i] = blo[i]; bF[tn][i + 8] = bhi[i]; }
    }

    // ---- issue next K-slab global loads early (overlap with WMMA) ----
    uint4 va0 = make_uint4(0u, 0u, 0u, 0u), va1 = va0, vb0 = va0, vb1 = va0;
    if (ki + 1 < KITERS) {
      va0 = *(const uint4*)(ag0 + kn);
      va1 = *(const uint4*)(ag1 + kn);
      vb0 = *(const uint4*)(bg0 + kn);
      vb1 = *(const uint4*)(bg1 + kn);
    }

    // ---- 8 WMMAs on the current slab ----
#pragma unroll
    for (int tm = 0; tm < 4; ++tm)
#pragma unroll
      for (int tn = 0; tn < 2; ++tn)
        acc[tm][tn] = __builtin_amdgcn_wmma_f32_16x16x32_f16(
            false, aF[tm], false, bF[tn], (short)0, acc[tm][tn], false, false);

    // ---- commit next slab to LDS buffer `nxt` ----
    if (ki + 1 < KITERS) {
      *(uint4*)&As[nxt][r0][o8] = va0;
      *(uint4*)&As[nxt][r1][o8] = va1;
      *(uint4*)&Bs[nxt][r0][o8] = vb0;
      *(uint4*)&Bs[nxt][r1][o8] = vb1;
    }
  }

  // ---- epilogue: D layout VGPR e -> M = e (+8 for lanes>=16), lane -> N ----
#pragma unroll
  for (int tm = 0; tm < 4; ++tm)
#pragma unroll
    for (int tn = 0; tn < 2; ++tn) {
      int col = colBase + waveN * 32 + tn * 16 + lm;
      if (col >= S) continue;
#pragma unroll
      for (int e = 0; e < 8; ++e) {
        int row = rowBase + waveM * 64 + tm * 16 + hs * 8 + e;
        if (row < L)
          simOut[((size_t)nb * L + row) * S + col] = acc[tm][tn][e] * postScale;
      }
    }
}

// -------------------------------------------------------------------------
// 3) per-row softmax stats (max, sum exp) over axis=2
// -------------------------------------------------------------------------
__global__ void row_softmax_stats(const float* __restrict__ sim,
                                  float* __restrict__ rmax, float* __restrict__ rsum,
                                  int S) {
  int rowg = blockIdx.x;
  const float* p = sim + (size_t)rowg * S;
  __shared__ float red[256];
  float m = -3.4e38f;
  for (int s = threadIdx.x; s < S; s += 256) m = fmaxf(m, p[s]);
  red[threadIdx.x] = m; __syncthreads();
  for (int st = 128; st > 0; st >>= 1) {
    if (threadIdx.x < st) red[threadIdx.x] = fmaxf(red[threadIdx.x], red[threadIdx.x + st]);
    __syncthreads();
  }
  m = red[0]; __syncthreads();
  float sum = 0.f;
  for (int s = threadIdx.x; s < S; s += 256) sum += __expf(p[s] - m);
  red[threadIdx.x] = sum; __syncthreads();
  for (int st = 128; st > 0; st >>= 1) {
    if (threadIdx.x < st) red[threadIdx.x] += red[threadIdx.x + st];
    __syncthreads();
  }
  if (threadIdx.x == 0) { rmax[rowg] = m; rsum[rowg] = red[0]; }
}

// -------------------------------------------------------------------------
// 4) per-column softmax stats over axis=1 (conf matrix is L2-resident: 184MB < 192MB)
// -------------------------------------------------------------------------
__global__ void col_softmax_stats(const float* __restrict__ sim,
                                  float* __restrict__ cmax, float* __restrict__ csum,
                                  int L, int S) {
  int colg = blockIdx.x;               // n*S + s
  int nb = colg / S, s = colg % S;
  const float* p = sim + (size_t)nb * L * S + s;
  __shared__ float red[256];
  float m = -3.4e38f;
  for (int r = threadIdx.x; r < L; r += 256) m = fmaxf(m, p[(size_t)r * S]);
  red[threadIdx.x] = m; __syncthreads();
  for (int st = 128; st > 0; st >>= 1) {
    if (threadIdx.x < st) red[threadIdx.x] = fmaxf(red[threadIdx.x], red[threadIdx.x + st]);
    __syncthreads();
  }
  m = red[0]; __syncthreads();
  float sum = 0.f;
  for (int r = threadIdx.x; r < L; r += 256) sum += __expf(p[(size_t)r * S] - m);
  red[threadIdx.x] = sum; __syncthreads();
  for (int st = 128; st > 0; st >>= 1) {
    if (threadIdx.x < st) red[threadIdx.x] += red[threadIdx.x + st];
    __syncthreads();
  }
  if (threadIdx.x == 0) { cmax[colg] = m; csum[colg] = red[0]; }
}

// -------------------------------------------------------------------------
// 5) conf = softmax1 * softmax2 (in place) + per-row conf max
// -------------------------------------------------------------------------
__global__ void finalize_conf(float* __restrict__ conf,
                              const float* __restrict__ rmax, const float* __restrict__ rsum,
                              const float* __restrict__ cmax, const float* __restrict__ csum,
                              float* __restrict__ rowConfMax, int L, int S) {
  int rowg = blockIdx.x;
  int nb = rowg / L;
  float* p = conf + (size_t)rowg * S;
  float rm = rmax[rowg], ri = 1.0f / rsum[rowg];
  __shared__ float red[256];
  float mx = 0.f;
  for (int s = threadIdx.x; s < S; s += 256) {
    float x = p[s];
    float c = __expf(x - rm) * ri * __expf(x - cmax[nb * S + s]) / csum[nb * S + s];
    p[s] = c;
    mx = fmaxf(mx, c);
  }
  red[threadIdx.x] = mx; __syncthreads();
  for (int st = 128; st > 0; st >>= 1) {
    if (threadIdx.x < st) red[threadIdx.x] = fmaxf(red[threadIdx.x], red[threadIdx.x + st]);
    __syncthreads();
  }
  if (threadIdx.x == 0) rowConfMax[rowg] = red[0];
}

// -------------------------------------------------------------------------
// 6) per-column conf max (L2-resident strided pass)
// -------------------------------------------------------------------------
__global__ void col_conf_max(const float* __restrict__ conf,
                             float* __restrict__ colConfMax, int L, int S) {
  int colg = blockIdx.x;
  int nb = colg / S, s = colg % S;
  const float* p = conf + (size_t)nb * L * S + s;
  __shared__ float red[256];
  float m = 0.f;
  for (int r = threadIdx.x; r < L; r += 256) m = fmaxf(m, p[(size_t)r * S]);
  red[threadIdx.x] = m; __syncthreads();
  for (int st = 128; st > 0; st >>= 1) {
    if (threadIdx.x < st) red[threadIdx.x] = fmaxf(red[threadIdx.x], red[threadIdx.x + st]);
    __syncthreads();
  }
  if (threadIdx.x == 0) colConfMax[colg] = red[0];
}

// -------------------------------------------------------------------------
// 7) threshold + border + mutual-NN + first-True extraction
// -------------------------------------------------------------------------
__global__ void extract_matches(const float* __restrict__ conf,
                                const float* __restrict__ rowConfMax,
                                const float* __restrict__ colConfMax,
                                float* __restrict__ out_maskv, float* __restrict__ out_j,
                                float* __restrict__ out_kpts, float* __restrict__ out_mconf,
                                int L, int S, int h0c, int w0c, int h1c, int w1c) {
  int rowg = blockIdx.x;
  int nb = rowg / L, l = rowg % L;
  const float* p = conf + (size_t)rowg * S;
  float rcm = rowConfMax[rowg];
  int y0 = l / w0c, x0 = l % w0c;
  bool rowValid = (y0 >= BORDER_C) && (y0 < h0c - BORDER_C) &&
                  (x0 >= BORDER_C) && (x0 < w0c - BORDER_C);
  __shared__ int redi[256];
  int best = 0x7fffffff;
  if (rowValid) {
    for (int s = threadIdx.x; s < S; s += 256) {
      float c = p[s];
      int y1 = s / w1c, x1 = s % w1c;
      bool ok = (c > THR_C) &&
                (y1 >= BORDER_C) && (y1 < h1c - BORDER_C) &&
                (x1 >= BORDER_C) && (x1 < w1c - BORDER_C) &&
                (c == rcm) && (c == colConfMax[nb * S + s]);
      if (ok && s < best) best = s;
    }
  }
  redi[threadIdx.x] = best; __syncthreads();
  for (int st = 128; st > 0; st >>= 1) {
    if (threadIdx.x < st) redi[threadIdx.x] = min(redi[threadIdx.x], redi[threadIdx.x + st]);
    __syncthreads();
  }
  if (threadIdx.x == 0) {
    int j = redi[0];
    bool has = (j != 0x7fffffff);
    if (!has) j = 0;                         // jnp.argmax of all-False -> 0
    out_maskv[rowg] = has ? 1.f : 0.f;
    out_j[rowg] = (float)j;
    out_kpts[(size_t)rowg * 2 + 0] = (float)(j % w1c);
    out_kpts[(size_t)rowg * 2 + 1] = (float)(j / w1c);
    out_mconf[rowg] = has ? p[j] : 0.f;
  }
}

// -------------------------------------------------------------------------
extern "C" void kernel_launch(void* const* d_in, const int* in_sizes, int n_in,
                              void* d_out, int out_size, void* d_ws, size_t ws_size,
                              hipStream_t stream) {
  const int h0c = 60, w0c = 80, h1c = 60, w1c = 80;   // from setup_inputs()
  const int L = h0c * w0c, S = h1c * w1c, C = 256;
  const int N = in_sizes[0] / (L * C);                // 2

  const float* f0 = (const float*)d_in[0];
  const float* f1 = (const float*)d_in[1];

  const int nA = in_sizes[0], nB = in_sizes[1];
  _Float16* a16 = (_Float16*)d_ws;
  _Float16* b16 = a16 + nA;
  float* stats = (float*)((char*)d_ws + ((size_t)(nA + nB) * 2 + 15) / 16 * 16);
  float* rmax = stats;                 // N*L
  float* rsum = rmax + (size_t)N * L;  // N*L
  float* cmax = rsum + (size_t)N * L;  // N*S
  float* csum = cmax + (size_t)N * S;  // N*S
  float* rcm  = csum + (size_t)N * S;  // N*L (row conf max)
  float* ccm  = rcm  + (size_t)N * L;  // N*S (col conf max)

  float* conf = (float*)d_out;                        // [N][L][S]
  size_t NLS = (size_t)N * L * S;
  float* out_maskv = conf + NLS;                      // N*L
  float* out_j     = out_maskv + (size_t)N * L;       // N*L
  float* out_kpts  = out_j + (size_t)N * L;           // N*L*2
  float* out_mconf = out_kpts + (size_t)2 * N * L;    // N*L

  // 1) convert + pre-scale by 1/16 (scale = C^-0.5)
  cvt_f16_scale<<<dim3((nA + 255) / 256), dim3(256), 0, stream>>>(f0, a16, nA, 0.0625f);
  cvt_f16_scale<<<dim3((nB + 255) / 256), dim3(256), 0, stream>>>(f1, b16, nB, 0.0625f);

  // 2) WMMA GEMM: sim = (a16 . b16) * 10  (10 = 1/TEMP)
  dim3 gGrid((S + 127) / 128, (L + 127) / 128, N);
  sim_gemm_wmma<<<gGrid, dim3(256), 0, stream>>>(a16, b16, conf, L, S, 10.0f);

  // 3-4) dual softmax stats
  row_softmax_stats<<<dim3(N * L), dim3(256), 0, stream>>>(conf, rmax, rsum, S);
  col_softmax_stats<<<dim3(N * S), dim3(256), 0, stream>>>(conf, cmax, csum, L, S);

  // 5) conf in place + row conf max
  finalize_conf<<<dim3(N * L), dim3(256), 0, stream>>>(conf, rmax, rsum, cmax, csum, rcm, L, S);

  // 6) col conf max
  col_conf_max<<<dim3(N * S), dim3(256), 0, stream>>>(conf, ccm, L, S);

  // 7) match extraction
  extract_matches<<<dim3(N * L), dim3(256), 0, stream>>>(conf, rcm, ccm,
      out_maskv, out_j, out_kpts, out_mconf, L, S, h0c, w0c, h1c, w1c);
}
